// CrossAttention_88776974008876
// MI455X (gfx1250) — matrix operand order repaired and checked by reading, not compiled, not run on hardware
//
#include <hip/hip_runtime.h>
#include <hip/hip_bf16.h>
#include <stdint.h>

typedef __attribute__((ext_vector_type(16))) __bf16 v16bf;
typedef __attribute__((ext_vector_type(8)))  float  v8f;

union VA { v16bf v; uint32_t u[8]; };
union VC { v8f   v; float    f[8]; };

__device__ __forceinline__ uint16_t f2bf(float f) {
    uint32_t u = __builtin_bit_cast(uint32_t, f);
    return (uint16_t)((u + 0x7FFFu + ((u >> 16) & 1u)) >> 16);
}

// Async DMA: global -> LDS, 16 bytes per lane, tracked by ASYNCcnt.
__device__ __forceinline__ void async_copy_b128(const void* g, void* l) {
    asm volatile("global_load_async_to_lds_b128 %0, %1, off"
                 :: "v"((uint32_t)(uintptr_t)l), "v"((uint64_t)(uintptr_t)g)
                 : "memory");
}
template <int N>
__device__ __forceinline__ void wait_asynccnt() {
    asm volatile("s_wait_asynccnt %0" :: "n"(N) : "memory");
}

// ---------------------------------------------------------------- cast kernel
__global__ void cast_f32_bf16_v4(const float* __restrict__ in,
                                 uint16_t* __restrict__ out, int n4) {
    int i = blockIdx.x * blockDim.x + threadIdx.x;
    if (i < n4) {
        float4 f = ((const float4*)in)[i];
        union { uint16_t h[4]; uint64_t u; } p;
        p.h[0] = f2bf(f.x); p.h[1] = f2bf(f.y);
        p.h[2] = f2bf(f.z); p.h[3] = f2bf(f.w);
        ((uint64_t*)out)[i] = p.u;
    }
}

// ---------------------------------------------------------------- GEMM
// C[M,N] = A[M,K] (bf16) @ W[N,K]^T (bf16). Block tile 128x128, 8 waves (2x4),
// wave tile 64x32. K-tile 32, double-buffered LDS fed by async DMA.
template <bool F32OUT>
__global__ __launch_bounds__(256)
void gemm_wmma_bf16(const uint16_t* __restrict__ A,
                    const uint16_t* __restrict__ W,
                    void* __restrict__ out,
                    const float* __restrict__ bias,
                    int M, int N, int K) {
    constexpr int LDT = 40;              // padded LDS row stride (elements)
    __shared__ __align__(16) uint16_t As[2][128 * LDT];
    __shared__ __align__(16) uint16_t Bs[2][128 * LDT];

    const int tid  = threadIdx.x;
    const int lane = tid & 31, wid = tid >> 5;
    const int wm = wid >> 2, wn = wid & 3;       // wave grid 2x4
    const int ln = lane & 15, hi = lane >> 4;
    const int m0 = blockIdx.y * 128, n0 = blockIdx.x * 128;

    VC c[4][2];
#pragma unroll
    for (int mf = 0; mf < 4; ++mf)
#pragma unroll
        for (int nf = 0; nf < 2; ++nf)
#pragma unroll
            for (int r = 0; r < 8; ++r) c[mf][nf].f[r] = 0.0f;

    auto stage = [&](int buf, int k0) {
#pragma unroll
        for (int i = 0; i < 2; ++i) {
            int flat = i * 256 + tid;            // 0..511, 4 uint4 per row
            int row = flat >> 2, c4 = flat & 3;
            async_copy_b128(A + (size_t)(m0 + row) * K + k0 + c4 * 8,
                            &As[buf][row * LDT + c4 * 8]);
            async_copy_b128(W + (size_t)(n0 + row) * K + k0 + c4 * 8,
                            &Bs[buf][row * LDT + c4 * 8]);
        }
    };

    const int T = K / 32;
    stage(0, 0);                                 // preload tile 0
    int buf = 0;
    for (int t = 0; t < T; ++t) {
        __syncthreads();                         // prev compute done everywhere
        if (t + 1 < T) {
            stage(buf ^ 1, (t + 1) * 32);        // prefetch next tile
            wait_asynccnt<4>();                  // current tile's 4 DMAs done
        } else {
            wait_asynccnt<0>();
        }
        __syncthreads();                         // tile visible to all waves

        const uint32_t* As32 = (const uint32_t*)As[buf];
        const uint32_t* Bs32 = (const uint32_t*)Bs[buf];
        VA a[4], bfr[2];
#pragma unroll
        for (int mf = 0; mf < 4; ++mf) {
            int m = wm * 64 + mf * 16 + ln;
#pragma unroll
            for (int r = 0; r < 8; ++r) {
                int k = (r < 4 ? 2 * r : 16 + 2 * (r - 4)) + (hi ? 8 : 0);
                a[mf].u[r] = As32[m * (LDT / 2) + (k >> 1)];
            }
        }
#pragma unroll
        for (int nf = 0; nf < 2; ++nf) {
            int n = wn * 32 + nf * 16 + ln;
#pragma unroll
            for (int r = 0; r < 8; ++r) {
                int k = (r < 4 ? 2 * r : 16 + 2 * (r - 4)) + (hi ? 8 : 0);
                bfr[nf].u[r] = Bs32[n * (LDT / 2) + (k >> 1)];
            }
        }
#pragma unroll
        for (int mf = 0; mf < 4; ++mf)
#pragma unroll
            for (int nf = 0; nf < 2; ++nf)
                c[mf][nf].v = __builtin_amdgcn_wmma_f32_16x16x32_bf16(
                    false, a[mf].v, false, bfr[nf].v,
                    (short)0, c[mf][nf].v, false, false);
        buf ^= 1;
    }

#pragma unroll
    for (int mf = 0; mf < 4; ++mf)
#pragma unroll
        for (int nf = 0; nf < 2; ++nf)
#pragma unroll
            for (int r = 0; r < 8; ++r) {
                int row = m0 + wm * 64 + mf * 16 + r + hi * 8;
                int col = n0 + wn * 32 + nf * 16 + ln;
                if (F32OUT)
                    ((float*)out)[(size_t)row * N + col] = c[mf][nf].f[r] + bias[col];
                else
                    ((uint16_t*)out)[(size_t)row * N + col] = f2bf(c[mf][nf].f[r]);
            }
}

// ---------------------------------------------------------------- flash attention
// One block = one (b, h, 64-query tile); 4 waves x 16 query rows.
// head_dim = 256, kv tile = 32 keys, online softmax, WMMA throughout.
// K/V tiles DMA'd to LDS; V wait deferred past softmax.
__global__ __launch_bounds__(128)
void flash_attn(const uint16_t* __restrict__ qp, const uint16_t* __restrict__ kp,
                const uint16_t* __restrict__ vp, uint16_t* __restrict__ ctx) {
    constexpr int LQ = 2048, LKV = 2048, D = 256, DM = 1024;
    constexpr int KT = 264;                          // padded K/V row stride
    __shared__ __align__(16) uint16_t Kt[32 * KT];
    __shared__ __align__(16) uint16_t Vt[32 * KT];
    __shared__ __align__(16) uint16_t Pb[4][16 * 34];

    const int blk = blockIdx.x;
    const int qtile = blk & 31, bh = blk >> 5;
    const int h = bh & 3, b = bh >> 2;
    const int q0 = qtile * 64;
    const int tid = threadIdx.x, lane = tid & 31, wid = tid >> 5;
    const int ln = lane & 15, hi = lane >> 4;

    // Q fragments for this wave's 16 rows (A layout, 8 x K=32 chunks of d)
    VA qf[8];
    {
        int qrow = q0 + wid * 16 + ln;
        const uint32_t* qg =
            (const uint32_t*)(qp + (size_t)(b * LQ + qrow) * DM + h * D);
#pragma unroll
        for (int j = 0; j < 8; ++j)
#pragma unroll
            for (int r = 0; r < 8; ++r) {
                int d = j * 32 + (r < 4 ? 2 * r : 16 + 2 * (r - 4)) + (hi ? 8 : 0);
                qf[j].u[r] = qg[d >> 1];
            }
    }

    VC ctxf[16];
#pragma unroll
    for (int c16 = 0; c16 < 16; ++c16)
#pragma unroll
        for (int r = 0; r < 8; ++r) ctxf[c16].f[r] = 0.0f;

    float Mr[8], Lr[8];
#pragma unroll
    for (int r = 0; r < 8; ++r) { Mr[r] = -INFINITY; Lr[r] = 0.0f; }
    const float scale = 0.0625f;   // 1/sqrt(256)

    const uint32_t* Kt32 = (const uint32_t*)Kt;
    const uint32_t* Pb32 = (const uint32_t*)&Pb[wid][0];

    for (int kt = 0; kt < LKV / 32; ++kt) {
        const int key0 = kt * 32;
        __syncthreads();                 // prev iter's Vt/Pb reads complete
        // DMA K tile first, then V tile (16 async ops per wave, in-order)
#pragma unroll
        for (int i = 0; i < 8; ++i) {
            int flat = i * 128 + tid;    // 0..1023, 32 uint4 per row
            int row = flat >> 5, c8 = flat & 31;
            async_copy_b128(kp + (size_t)(b * LKV + key0 + row) * DM + h * D + c8 * 8,
                            Kt + row * KT + c8 * 8);
        }
#pragma unroll
        for (int i = 0; i < 8; ++i) {
            int flat = i * 128 + tid;
            int row = flat >> 5, c8 = flat & 31;
            async_copy_b128(vp + (size_t)(b * LKV + key0 + row) * DM + h * D + c8 * 8,
                            Vt + row * KT + c8 * 8);
        }
        wait_asynccnt<8>();              // K tile landed (V still in flight)
        __syncthreads();

        // S = Q K^T : two 16x16 fragments (keys 0-15, 16-31)
        VC s0, s1;
#pragma unroll
        for (int r = 0; r < 8; ++r) { s0.f[r] = 0.0f; s1.f[r] = 0.0f; }
#pragma unroll
        for (int j = 0; j < 8; ++j) {
            VA b0, b1;
#pragma unroll
            for (int r = 0; r < 8; ++r) {
                int d = j * 32 + (r < 4 ? 2 * r : 16 + 2 * (r - 4)) + (hi ? 8 : 0);
                b0.u[r] = Kt32[ln * (KT / 2) + (d >> 1)];
                b1.u[r] = Kt32[(16 + ln) * (KT / 2) + (d >> 1)];
            }
            s0.v = __builtin_amdgcn_wmma_f32_16x16x32_bf16(
                false, qf[j].v, false, b0.v, (short)0, s0.v, false, false);
            s1.v = __builtin_amdgcn_wmma_f32_16x16x32_bf16(
                false, qf[j].v, false, b1.v, (short)0, s1.v, false, false);
        }

        // online softmax (row m = r + 8*hi lives across 16 lanes of a half-wave)
        float p0[8], p1[8], alpha[8];
#pragma unroll
        for (int r = 0; r < 8; ++r) {
            float v0 = s0.f[r] * scale, v1 = s1.f[r] * scale;
            float mx = fmaxf(v0, v1);
#pragma unroll
            for (int msk = 1; msk < 16; msk <<= 1)
                mx = fmaxf(mx, __shfl_xor(mx, msk, 32));
            float Mn = fmaxf(Mr[r], mx);
            alpha[r] = __expf(Mr[r] - Mn);
            Mr[r] = Mn;
            p0[r] = __expf(v0 - Mn);
            p1[r] = __expf(v1 - Mn);
            float sum = p0[r] + p1[r];
#pragma unroll
            for (int msk = 1; msk < 16; msk <<= 1)
                sum += __shfl_xor(sum, msk, 32);
            Lr[r] = Lr[r] * alpha[r] + sum;
        }
#pragma unroll
        for (int c16 = 0; c16 < 16; ++c16)
#pragma unroll
            for (int r = 0; r < 8; ++r) ctxf[c16].f[r] *= alpha[r];

        // re-layout P: C-fragment -> LDS -> A-fragment
#pragma unroll
        for (int r = 0; r < 8; ++r) {
            int m = r + hi * 8;
            Pb[wid][m * 34 + ln]      = f2bf(p0[r]);
            Pb[wid][m * 34 + 16 + ln] = f2bf(p1[r]);
        }
        wait_asynccnt<0>();              // V tile landed
        __syncthreads();                 // V + P visible everywhere

        VA pa;
#pragma unroll
        for (int r = 0; r < 8; ++r) {
            int k = (r < 4 ? 2 * r : 16 + 2 * (r - 4)) + (hi ? 8 : 0);
            pa.u[r] = Pb32[ln * 17 + (k >> 1)];
        }
        // ctx += P V : 16 WMMAs over the 256-wide head dim
#pragma unroll
        for (int c16 = 0; c16 < 16; ++c16) {
            VA vb;
            int n = c16 * 16 + ln;
#pragma unroll
            for (int r = 0; r < 8; ++r) {
                int k = (r < 4 ? 2 * r : 16 + 2 * (r - 4)) + (hi ? 8 : 0);
                uint32_t lo = Vt[k * KT + n];
                uint32_t h2 = Vt[(k + 1) * KT + n];
                vb.u[r] = lo | (h2 << 16);
            }
            ctxf[c16].v = __builtin_amdgcn_wmma_f32_16x16x32_bf16(
                false, pa.v, false, vb.v, (short)0, ctxf[c16].v, false, false);
        }
    }

    // epilogue: normalize and store bf16 ctx at [b, q, h*256 + d]
#pragma unroll
    for (int r = 0; r < 8; ++r) Lr[r] = 1.0f / Lr[r];
    const int orow = q0 + wid * 16;
#pragma unroll
    for (int c16 = 0; c16 < 16; ++c16)
#pragma unroll
        for (int r = 0; r < 8; ++r) {
            int row = orow + r + hi * 8;
            int col = h * D + c16 * 16 + ln;
            ctx[(size_t)(b * LQ + row) * DM + col] = f2bf(ctxf[c16].f[r] * Lr[r]);
        }
}

// ---------------------------------------------------------------- launcher
extern "C" void kernel_launch(void* const* d_in, const int* in_sizes, int n_in,
                              void* d_out, int out_size, void* d_ws, size_t ws_size,
                              hipStream_t stream) {
    const float* q  = (const float*)d_in[0];
    const float* kv = (const float*)d_in[1];
    const float* Wq = (const float*)d_in[2];
    const float* Wk = (const float*)d_in[3];
    const float* Wv = (const float*)d_in[4];
    const float* Wo = (const float*)d_in[5];
    const float* bo = (const float*)d_in[6];
    float* out = (float*)d_out;

    constexpr int B = 4, LQ = 2048, LKV = 2048, DQ = 1024, DKV = 768;
    const int Mq = B * LQ;       // 8192
    const int Mk = B * LKV;      // 8192

    char* ws = (char*)d_ws;
    size_t off = 0;
    auto alloc = [&](size_t bytes) -> void* {
        void* p = ws + off;
        off = (off + bytes + 255) & ~(size_t)255;
        return p;
    };
    uint16_t* qb  = (uint16_t*)alloc((size_t)Mq * DQ * 2);
    uint16_t* kvb = (uint16_t*)alloc((size_t)Mk * DKV * 2);
    uint16_t* Wqb = (uint16_t*)alloc((size_t)DQ * DQ * 2);
    uint16_t* Wkb = (uint16_t*)alloc((size_t)DQ * DKV * 2);
    uint16_t* Wvb = (uint16_t*)alloc((size_t)DQ * DKV * 2);
    uint16_t* Wob = (uint16_t*)alloc((size_t)DQ * DQ * 2);
    uint16_t* qp  = (uint16_t*)alloc((size_t)Mq * DQ * 2);
    uint16_t* kp  = (uint16_t*)alloc((size_t)Mk * DQ * 2);
    uint16_t* vp  = (uint16_t*)alloc((size_t)Mk * DQ * 2);
    uint16_t* ctx = (uint16_t*)alloc((size_t)Mq * DQ * 2);

    auto cast = [&](const float* src, uint16_t* dst, int n) {
        cast_f32_bf16_v4<<<(n / 4 + 255) / 256, 256, 0, stream>>>(src, dst, n / 4);
    };
    cast(q,  qb,  Mq * DQ);
    cast(kv, kvb, Mk * DKV);
    cast(Wq, Wqb, DQ * DQ);
    cast(Wk, Wkb, DQ * DKV);
    cast(Wv, Wvb, DQ * DKV);
    cast(Wo, Wob, DQ * DQ);

    dim3 gproj(DQ / 128, Mq / 128);          // (8, 64)
    gemm_wmma_bf16<false><<<gproj, 256, 0, stream>>>(qb,  Wqb, qp, nullptr, Mq, DQ, DQ);
    gemm_wmma_bf16<false><<<gproj, 256, 0, stream>>>(kvb, Wkb, kp, nullptr, Mk, DQ, DKV);
    gemm_wmma_bf16<false><<<gproj, 256, 0, stream>>>(kvb, Wvb, vp, nullptr, Mk, DQ, DKV);

    flash_attn<<<B * 4 * (LQ / 64), 128, 0, stream>>>(qp, kp, vp, ctx);

    gemm_wmma_bf16<true><<<gproj, 256, 0, stream>>>(ctx, Wob, out, bo, Mq, DQ, DQ);
}
